// CMHSA_17557826306152
// MI455X (gfx1250) — matrix operand order repaired and checked by reading, not compiled, not run on hardware
//
#include <hip/hip_runtime.h>
#include <hip/hip_bf16.h>

typedef __attribute__((ext_vector_type(16))) __bf16       v16bf;
typedef __attribute__((ext_vector_type(8)))  float        v8f;
typedef __attribute__((ext_vector_type(4)))  unsigned int v4u;
typedef __attribute__((ext_vector_type(8)))  unsigned int v8u;

#define BATCH 2
#define CCH   256
#define NH    8
#define HD    32
#define NTOK  4096
#define SCALE 0.17677669529663687f  // 1/sqrt(32)

__device__ __forceinline__ unsigned short f2bf(float f) {
    unsigned int u = __builtin_bit_cast(unsigned int, f);
    u += 0x7FFFu + ((u >> 16) & 1u);           // round-to-nearest-even
    return (unsigned short)(u >> 16);
}

// ---------------------------------------------------------------- K0: weights fp32 -> bf16
__global__ __launch_bounds__(256) void k_cvt_w(const float* __restrict__ Wq,
                                               const float* __restrict__ Wk,
                                               const float* __restrict__ Wv,
                                               const float* __restrict__ Wo,
                                               unsigned short* __restrict__ dst) {
    int i = blockIdx.x * 256 + threadIdx.x;    // 65536 total
    dst[i]           = f2bf(Wq[i]);
    dst[65536  + i]  = f2bf(Wk[i]);
    dst[131072 + i]  = f2bf(Wv[i]);
    dst[196608 + i]  = f2bf(Wo[i]);
}

// ---------------------------------------------------------------- K1: fused QKV projection
__global__ __launch_bounds__(256) void k_qkv(const float* __restrict__ x,
                                             const unsigned short* __restrict__ wbf,
                                             const float* __restrict__ bq,
                                             const float* __restrict__ bk,
                                             const float* __restrict__ bv,
                                             unsigned short* __restrict__ qws,
                                             unsigned short* __restrict__ kws,
                                             unsigned short* __restrict__ vws) {
    __shared__ unsigned short xs[16 * 264];    // 16 tokens x 256ch, row stride 528B (conflict-free)
    const int blk = blockIdx.x;                // 512 blocks
    const int b   = blk >> 8;
    const int t0  = (blk & 255) * 16;
    const int tid = threadIdx.x;

    {   // cooperative load: thread 'c' loads 16 tokens of channel c (contiguous in x)
        const int c = tid;
        const float* src = x + ((size_t)b * CCH + c) * NTOK + t0;
        #pragma unroll
        for (int t4 = 0; t4 < 4; t4++) {
            float4 fv = *(const float4*)(src + 4 * t4);
            xs[(4 * t4 + 0) * 264 + c] = f2bf(fv.x);
            xs[(4 * t4 + 1) * 264 + c] = f2bf(fv.y);
            xs[(4 * t4 + 2) * 264 + c] = f2bf(fv.z);
            xs[(4 * t4 + 3) * 264 + c] = f2bf(fv.w);
        }
    }
    __syncthreads();

    const int lane = tid & 31, wid = tid >> 5;
    const int lr = lane & 15, hi = lane >> 4;

    for (int j = 0; j < 6; j++) {
        const int job = wid + 8 * j;           // 48 jobs: 3 mats x 16 och-tiles
        const int mat = job >> 4;
        const int o0  = (job & 15) * 16;
        const unsigned short* wmat = wbf + mat * 65536;
        const float* bias = (mat == 0) ? bq : ((mat == 1) ? bk : bv);

        v8f acc = {};
        #pragma unroll
        for (int kb = 0; kb < 8; kb++) {
            const int c0 = kb * 32;
            const unsigned short* arow = xs + lr * 264 + c0 + (hi ? 16 : 0);
            union { v4u h[2]; v16bf v; } au;
            au.h[0] = *(const v4u*)(arow);
            au.h[1] = *(const v4u*)(arow + 8);
            v8u bbu = *(const v8u*)(wmat + (size_t)(o0 + lr) * 256 + c0 + (hi ? 16 : 0));
            v16bf bb = __builtin_bit_cast(v16bf, bbu);
            acc = __builtin_amdgcn_wmma_f32_16x16x32_bf16(false, au.v, false, bb,
                                                          (short)0, acc, false, false);
        }
        const int och  = o0 + lr;
        const int head = och >> 5;
        const int dl   = och & 31;
        const float bsv = bias[och];
        if (mat < 2) {  // q (pre-scaled) and k -> [b,h,n,d]
            unsigned short* dstp = (mat == 0) ? qws : kws;
            const float sc = (mat == 0) ? SCALE : 1.0f;
            #pragma unroll
            for (int r = 0; r < 8; r++) {
                const int tok = t0 + r + 8 * hi;
                dstp[(((size_t)b * NH + head) * NTOK + tok) * HD + dl] = f2bf((acc[r] + bsv) * sc);
            }
        } else {        // v -> transposed [b,h,d,n]
            #pragma unroll
            for (int r = 0; r < 8; r++) {
                const int tok = t0 + r + 8 * hi;
                vws[(((size_t)b * NH + head) * HD + dl) * NTOK + tok] = f2bf(acc[r] + bsv);
            }
        }
    }
}

// ---------------------------------------------------------------- K2: flash attention
// 8 waves/block, all sharing the same (batch,head); K/V tiles (64 keys) are
// double-buffered in LDS via async-to-LDS copies; each wave owns 16 q-rows.
__global__ __launch_bounds__(256) void k_attn(const unsigned short* __restrict__ qws,
                                              const unsigned short* __restrict__ kws,
                                              const unsigned short* __restrict__ vws,
                                              unsigned short* __restrict__ ao) {
    __shared__ unsigned short klds[2][64 * 40]; // K stage: 64 keys x 32d, row stride 80B
    __shared__ unsigned short vlds[2][32 * 72]; // V stage: 32 d x 64 keys, row stride 144B
    __shared__ unsigned short ps[8 * 16 * 40];  // per-wave 16x32 P transpose, row stride 80B

    const int tid  = threadIdx.x;
    const int lane = tid & 31, wid = tid >> 5;
    const int lr = lane & 15, hi = lane >> 4;
    const int gw = blockIdx.x * 8 + wid;        // 4096 row-block waves
    const int bh = blockIdx.x >> 5;             // uniform per workgroup
    const int t0 = (gw & 255) * 16;

    const unsigned short* qbase = qws + (size_t)bh * NTOK * HD;
    const unsigned short* kbase = kws + (size_t)bh * NTOK * HD;
    const unsigned short* vbase = vws + (size_t)bh * HD * NTOK;
    unsigned short* myps = ps + wid * 640;

    // cooperative staging roles (256 threads, 16B per thread per tile)
    const int krow = tid >> 2, kcol = tid & 3;  // K: 64 rows x 4 chunks
    const int vrow = tid >> 3, vcol = tid & 7;  // V: 32 rows x 8 chunks
    const unsigned kbuf0 = (unsigned)(size_t)&klds[0][0];
    const unsigned kbuf1 = (unsigned)(size_t)&klds[1][0];
    const unsigned vbuf0 = (unsigned)(size_t)&vlds[0][0];
    const unsigned vbuf1 = (unsigned)(size_t)&vlds[1][0];
    const unsigned kdsto = (unsigned)(krow * 80 + kcol * 16);
    const unsigned vdsto = (unsigned)(vrow * 144 + vcol * 16);
    const int kgo = krow * 64 + kcol * 16;      // + k0*64 (bytes)
    const int vgo = vrow * 8192 + vcol * 16;    // + k0*2  (bytes)

    // prologue: stage key-block 0 into buffer 0
    asm volatile("global_load_async_to_lds_b128 %0, %1, %2"
                 :: "v"(kbuf0 + kdsto), "v"(kgo), "s"(kbase) : "memory");
    asm volatile("global_load_async_to_lds_b128 %0, %1, %2"
                 :: "v"(vbuf0 + vdsto), "v"(vgo), "s"(vbase) : "memory");

    union U { v4u h[2]; v16bf v; };
    // q A-tile (already scaled by d^-1/2)
    const unsigned short* qrow = qbase + (size_t)(t0 + lr) * HD + (hi ? 16 : 0);
    U qa; qa.h[0] = *(const v4u*)qrow; qa.h[1] = *(const v4u*)(qrow + 8);

    float m[8], l[8];
    v8f o0acc = {}, o1acc = {};
    #pragma unroll
    for (int r = 0; r < 8; r++) { m[r] = -1e30f; l[r] = 0.f; }

    for (int kb = 0; kb < 64; kb++) {           // 64 stages of 64 keys
        const int buf = kb & 1;
        asm volatile("s_wait_asynccnt 0x0" ::: "memory");
        __syncthreads();                        // current stage visible to all waves
        if (kb + 1 < 64) {                      // kick next stage into the other buffer
            const int k0n = (kb + 1) * 64;
            const unsigned kd = (buf ? kbuf0 : kbuf1) + kdsto;
            const unsigned vd = (buf ? vbuf0 : vbuf1) + vdsto;
            asm volatile("global_load_async_to_lds_b128 %0, %1, %2"
                         :: "v"(kd), "v"(kgo + k0n * 64), "s"(kbase) : "memory");
            asm volatile("global_load_async_to_lds_b128 %0, %1, %2"
                         :: "v"(vd), "v"(vgo + k0n * 2), "s"(vbase) : "memory");
        }
        const unsigned short* kc = buf ? &klds[1][0] : &klds[0][0];
        const unsigned short* vc = buf ? &vlds[1][0] : &vlds[0][0];

        #pragma unroll
        for (int s = 0; s < 2; s++) {           // two 32-key sub-blocks per stage
            // K^T B-tiles from LDS: column = key, K-dim = d
            v8u kb0u = *(const v8u*)(kc + (s * 32 + lr) * 40 + (hi ? 16 : 0));
            v8u kb1u = *(const v8u*)(kc + (s * 32 + 16 + lr) * 40 + (hi ? 16 : 0));
            v8f z = {};
            v8f s0 = __builtin_amdgcn_wmma_f32_16x16x32_bf16(false, qa.v, false,
                         __builtin_bit_cast(v16bf, kb0u), (short)0, z, false, false);
            v8f s1 = __builtin_amdgcn_wmma_f32_16x16x32_bf16(false, qa.v, false,
                         __builtin_bit_cast(v16bf, kb1u), (short)0, z, false, false);

            #pragma unroll
            for (int r = 0; r < 8; r++) {
                float t = fmaxf(s0[r], s1[r]);
                t = fmaxf(t, __shfl_xor(t, 1, 32));
                t = fmaxf(t, __shfl_xor(t, 2, 32));
                t = fmaxf(t, __shfl_xor(t, 4, 32));
                t = fmaxf(t, __shfl_xor(t, 8, 32));
                const float mn   = fmaxf(m[r], t);
                const float corr = __expf(m[r] - mn);
                m[r] = mn;
                const float p0 = __expf(s0[r] - mn);
                const float p1 = __expf(s1[r] - mn);
                float rs = p0 + p1;
                rs += __shfl_xor(rs, 1, 32);
                rs += __shfl_xor(rs, 2, 32);
                rs += __shfl_xor(rs, 4, 32);
                rs += __shfl_xor(rs, 8, 32);
                l[r] = l[r] * corr + rs;
                o0acc[r] *= corr;
                o1acc[r] *= corr;
                unsigned short* pr = myps + (r + 8 * hi) * 40 + lr;
                pr[0]  = f2bf(p0);
                pr[16] = f2bf(p1);
            }
            asm volatile("s_wait_dscnt 0" ::: "memory");  // per-wave DS in-order; cross-lane RAW fence
            const unsigned short* prow = myps + lr * 40 + (hi ? 16 : 0);
            U pa; pa.h[0] = *(const v4u*)prow; pa.h[1] = *(const v4u*)(prow + 8);
            // V B-tiles from LDS: column = d, K-dim = key
            v8u vb0u = *(const v8u*)(vc + lr * 72 + s * 32 + (hi ? 16 : 0));
            v8u vb1u = *(const v8u*)(vc + (16 + lr) * 72 + s * 32 + (hi ? 16 : 0));
            o0acc = __builtin_amdgcn_wmma_f32_16x16x32_bf16(false, pa.v, false,
                        __builtin_bit_cast(v16bf, vb0u), (short)0, o0acc, false, false);
            o1acc = __builtin_amdgcn_wmma_f32_16x16x32_bf16(false, pa.v, false,
                        __builtin_bit_cast(v16bf, vb1u), (short)0, o1acc, false, false);
        }
        __syncthreads();                        // all waves done with this buffer
    }

    // normalize and store attn output channel-major [b, h*32+d, n] (bf16)
    unsigned short* aob = ao + (size_t)bh * HD * NTOK;
    #pragma unroll
    for (int r = 0; r < 8; r++) {
        const float inv = 1.0f / l[r];
        const int tok = t0 + r + 8 * hi;
        aob[(size_t)lr * NTOK + tok]        = f2bf(o0acc[r] * inv);
        aob[(size_t)(16 + lr) * NTOK + tok] = f2bf(o1acc[r] * inv);
    }
}

// ---------------------------------------------------------------- K3: output projection + bias + residual
__global__ __launch_bounds__(256) void k_proj(const unsigned short* __restrict__ ao,
                                              const unsigned short* __restrict__ wbf,
                                              const float* __restrict__ bo,
                                              const float* __restrict__ x,
                                              float* __restrict__ out) {
    __shared__ unsigned short xs[16 * 264];
    const int blk = blockIdx.x;
    const int b   = blk >> 8;
    const int t0  = (blk & 255) * 16;
    const int tid = threadIdx.x;

    {   // cooperative load of attn-out tile (bf16, contiguous per channel)
        const int c = tid;
        const unsigned short* src = ao + ((size_t)b * CCH + c) * NTOK + t0;
        v8u dv = *(const v8u*)src;
        #pragma unroll
        for (int t = 0; t < 8; t++) {
            unsigned int w = dv[t];
            xs[(2 * t + 0) * 264 + c] = (unsigned short)(w & 0xffffu);
            xs[(2 * t + 1) * 264 + c] = (unsigned short)(w >> 16);
        }
    }
    __syncthreads();

    const int lane = tid & 31, wid = tid >> 5;
    const int lr = lane & 15, hi = lane >> 4;
    const unsigned short* wo = wbf + 3 * 65536;

    for (int j = 0; j < 2; j++) {
        const int o0 = (wid + 8 * j) * 16;
        v8f acc = {};
        #pragma unroll
        for (int kb = 0; kb < 8; kb++) {
            const int c0 = kb * 32;
            const unsigned short* arow = xs + lr * 264 + c0 + (hi ? 16 : 0);
            union U { v4u h[2]; v16bf v; } au;
            au.h[0] = *(const v4u*)(arow);
            au.h[1] = *(const v4u*)(arow + 8);
            v8u bbu = *(const v8u*)(wo + (size_t)(o0 + lr) * 256 + c0 + (hi ? 16 : 0));
            acc = __builtin_amdgcn_wmma_f32_16x16x32_bf16(false, au.v, false,
                      __builtin_bit_cast(v16bf, bbu), (short)0, acc, false, false);
        }
        const int och = o0 + lr;
        const float bsv = bo[och];
        const float* xr  = x   + ((size_t)b * CCH + och) * NTOK + t0;
        float*       orr = out + ((size_t)b * CCH + och) * NTOK + t0;
        #pragma unroll
        for (int r = 0; r < 8; r++) {
            const int tok = r + 8 * hi;
            orr[tok] = xr[tok] + bsv + acc[r];
        }
    }
}

// ---------------------------------------------------------------- host launcher
extern "C" void kernel_launch(void* const* d_in, const int* in_sizes, int n_in,
                              void* d_out, int out_size, void* d_ws, size_t ws_size,
                              hipStream_t stream) {
    const float* x  = (const float*)d_in[0];
    const float* Wq = (const float*)d_in[1];
    const float* bq = (const float*)d_in[2];
    const float* Wk = (const float*)d_in[3];
    const float* bk = (const float*)d_in[4];
    const float* Wv = (const float*)d_in[5];
    const float* bv = (const float*)d_in[6];
    const float* Wo = (const float*)d_in[7];
    const float* bo = (const float*)d_in[8];
    float* out = (float*)d_out;

    unsigned short* wbf = (unsigned short*)d_ws;          // 4 x 65536 bf16 weights
    unsigned short* qws = wbf + 4 * 65536;                // [2,8,4096,32] bf16
    unsigned short* kws = qws + 2097152;                  // [2,8,4096,32] bf16
    unsigned short* vws = kws + 2097152;                  // [2,8,32,4096] bf16
    unsigned short* ao  = vws + 2097152;                  // [2,256,4096]  bf16

    k_cvt_w<<<256, 256, 0, stream>>>(Wq, Wk, Wv, Wo, wbf);
    k_qkv  <<<512, 256, 0, stream>>>(x, wbf, bq, bk, bv, qws, kws, vws);
    k_attn <<<512, 256, 0, stream>>>(qws, kws, vws, ao);
    k_proj <<<512, 256, 0, stream>>>(ao, wbf, bo, x, out);
}
